// MultiHeadSelfAttention_77275051590175
// MI455X (gfx1250) — compile-verified, matrix operand order checked by reading, not run on hardware
//
#include <hip/hip_runtime.h>
#include <stdint.h>

typedef __attribute__((ext_vector_type(16))) __bf16 v16bf;
typedef __attribute__((ext_vector_type(8)))  float  v8f;
typedef unsigned int u32x4 __attribute__((ext_vector_type(4)));
typedef int          i32x8 __attribute__((ext_vector_type(8)));
typedef int          i32x4 __attribute__((ext_vector_type(4)));

namespace {

constexpr int Bsz = 512, Tsz = 64, Dsz = 1024, Hn = 16, DK = 64;
constexpr int Mtot = Bsz * Tsz;   // 32768
constexpr int Nqkv = 3 * Dsz;     // 3072
constexpr int BM = 128, BN = 128, BK = 64;
constexpr int KIT = Dsz / BK;     // 16
constexpr int LSTRIDE = 72;       // 144B rows: 128B data + 16B TDM pad -> conflict-free b128

struct U4x2 { uint4 lo, hi; };

__device__ __forceinline__ uint32_t f2bf(float f) {
  uint32_t u = __builtin_bit_cast(uint32_t, f);
  u += 0x7fffu + ((u >> 16) & 1u);   // round-to-nearest-even
  return u >> 16;
}

// 16-bit WMMA A/B fragment: elements 0-7 = K[lh*8..], elements 8-15 = K[16+lh*8..]
__device__ __forceinline__ v16bf load_frag(const uint16_t* rowbase, int lh) {
  U4x2 u;
  u.lo = *(const uint4*)(rowbase + lh * 8);
  u.hi = *(const uint4*)(rowbase + 16 + lh * 8);
  return __builtin_bit_cast(v16bf, u);
}

__device__ __forceinline__ v8f wmma_bf16(v16bf a, v16bf b, v8f c) {
  return __builtin_amdgcn_wmma_f32_16x16x32_bf16(false, a, false, b, (short)0, c, false, false);
}

__device__ __forceinline__ void set_bits(uint32_t* w, int lo, int len, uint64_t val) {
  int word = lo >> 5, sh = lo & 31;
  w[word] |= (uint32_t)((val << sh) & 0xffffffffull);
  if (sh + len > 32) w[word + 1] |= (uint32_t)(val >> (32 - sh));
}

// TDM: DMA a 2D bf16 tile (tile_w elems wide, tile_h rows, global row stride
// stride_elems) into LDS, padding +16B after every 128B so rows land at the
// conflict-free 144B LDS stride. Wave-level op; EXEC ignored; TENSORcnt tracked.
__device__ __forceinline__ void tdm_load_tile(const uint16_t* gsrc, uint32_t lds_addr,
                                              uint32_t tile_w, uint32_t tile_h,
                                              uint32_t stride_elems) {
  uint32_t g0[4] = {0, 0, 0, 0};
  uint32_t g1[8] = {0, 0, 0, 0, 0, 0, 0, 0};
  uint64_t ga = (uint64_t)(uintptr_t)gsrc;
  g0[0] = 1u;                                            // count = 1 (valid D#)
  g0[1] = lds_addr;                                      // LDS byte address
  g0[2] = (uint32_t)ga;                                  // global_addr[31:0]
  g0[3] = (uint32_t)((ga >> 32) & 0x01ffffffu) | (2u << 30);  // addr[56:32] | type=2
  set_bits(g1, 16, 2, 1);                 // data_size = 2 bytes
  set_bits(g1, 20, 1, 1);                 // pad_enable
  set_bits(g1, 22, 3, 4);                 // pad_interval code 4 = 32 DWORDs (128B)
  set_bits(g1, 25, 7, 3);                 // pad_amount  code 3 =  4 DWORDs (16B)
  set_bits(g1, 48, 32, 1u << 30);         // tensor_dim0 (huge: no clipping)
  set_bits(g1, 80, 32, 1u << 30);         // tensor_dim1
  set_bits(g1, 112, 16, tile_w);          // tile_dim0
  set_bits(g1, 128, 16, tile_h);          // tile_dim1
  set_bits(g1, 160, 48, stride_elems);    // tensor_dim0_stride (elements)
  u32x4 v0 = {g0[0], g0[1], g0[2], g0[3]};
  i32x8 v1 = {(int)g1[0], (int)g1[1], (int)g1[2], (int)g1[3],
              (int)g1[4], (int)g1[5], (int)g1[6], (int)g1[7]};
  i32x4 vz4 = {0, 0, 0, 0};
  i32x8 vz8 = {0, 0, 0, 0, 0, 0, 0, 0};
  __builtin_amdgcn_tensor_load_to_lds(v0, v1, vz4, vz4, vz8, 0);
}

// ---------------- prep kernels ----------------
__global__ void k_prep_x(const float* __restrict__ x, uint16_t* __restrict__ xb) {
  int idx = blockIdx.x * 256 + threadIdx.x;
  if (idx < Mtot * Dsz) xb[idx] = (uint16_t)f2bf(x[idx]);
}

__global__ void k_prep_wqkv(const float* __restrict__ Wq, const float* __restrict__ Wk,
                            const float* __restrict__ Wv, uint16_t* __restrict__ wt) {
  int idx = blockIdx.x * 256 + threadIdx.x;
  if (idx >= Nqkv * Dsz) return;
  int n = idx >> 10, k = idx & 1023;
  int g = n >> 10, n2 = n & 1023;
  const float* W = (g == 0) ? Wq : (g == 1) ? Wk : Wv;
  wt[idx] = (uint16_t)f2bf(W[k * Dsz + n2]);   // (N,K) = W^T
}

__global__ void k_prep_wo(const float* __restrict__ Wo, uint16_t* __restrict__ wt) {
  int idx = blockIdx.x * 256 + threadIdx.x;
  if (idx >= Dsz * Dsz) return;
  int n = idx >> 10, k = idx & 1023;
  wt[idx] = (uint16_t)f2bf(Wo[k * Dsz + n]);
}

__global__ void k_prep_bias(const float* __restrict__ bq, const float* __restrict__ bk,
                            const float* __restrict__ bv, float* __restrict__ bqkv) {
  int idx = blockIdx.x * 256 + threadIdx.x;
  if (idx >= Nqkv) return;
  int g = idx >> 10, n = idx & 1023;
  bqkv[idx] = (g == 0) ? bq[n] : (g == 1) ? bk[n] : bv[n];
}

__global__ void k_bias_tab(const float* __restrict__ rel, float* __restrict__ btab) {
  int idx = blockIdx.x * 256 + threadIdx.x;
  if (idx >= Hn * Tsz * Tsz) return;
  int h = idx >> 12, q = (idx >> 6) & 63, t = idx & 63;
  int dr = (q >> 3) - (t >> 3), df = (q & 7) - (t & 7);
  int ri = min(max(dr, -7), 7) + 7;
  int fi = min(max(df, -7), 7) + 7;
  btab[idx] = tanhf(rel[(ri * 15 + fi) * Hn + h]) * 2.0f;  // (h,q,t)
}

// ---------------- TDM-fed double-buffered WMMA GEMM ----------------
// MODE 0: epilogue scatters bf16 Q,K (b,h,t,d) and V^T (b,h,d,t).
// MODE 1: epilogue writes f32 out + bo.
template <int MODE>
__global__ __launch_bounds__(256) void k_gemm(
    const uint16_t* __restrict__ Ab, const uint16_t* __restrict__ Bt,
    const float* __restrict__ biasv,
    uint16_t* __restrict__ qws, uint16_t* __restrict__ kws,
    uint16_t* __restrict__ vtws, float* __restrict__ outp) {
  __shared__ __align__(16) uint16_t As[2][BM][LSTRIDE];
  __shared__ __align__(16) uint16_t Bs[2][BN][LSTRIDE];

  const int tid = threadIdx.x;
  const int wid = tid >> 5, lane = tid & 31;
  const int l = lane & 15, lh = lane >> 4;
  const int waveM = wid & 1;    // 2 waves in M -> 64 rows each
  const int waveN = wid >> 1;   // 4 waves in N -> 32 cols each
  const int mbase = blockIdx.y * BM;
  const int nbase = blockIdx.x * BN;

  const uint16_t* Abase = Ab + (size_t)mbase * Dsz;
  const uint16_t* Bbase = Bt + (size_t)nbase * Dsz;
  const uint32_t ldsA0 = (uint32_t)(uintptr_t)&As[0][0][0];
  const uint32_t ldsA1 = (uint32_t)(uintptr_t)&As[1][0][0];
  const uint32_t ldsB0 = (uint32_t)(uintptr_t)&Bs[0][0][0];
  const uint32_t ldsB1 = (uint32_t)(uintptr_t)&Bs[1][0][0];

  v8f acc[4][2];
  v8f zero = {0.f, 0.f, 0.f, 0.f, 0.f, 0.f, 0.f, 0.f};
#pragma unroll
  for (int mi = 0; mi < 4; ++mi)
#pragma unroll
    for (int ni = 0; ni < 2; ++ni) acc[mi][ni] = zero;

  if (wid == 0) {                       // prologue: stage slab 0
    tdm_load_tile(Abase, ldsA0, BK, BM, Dsz);
    tdm_load_tile(Bbase, ldsB0, BK, BN, Dsz);
  }

  for (int it = 0; it < KIT; ++it) {
    const int cur = it & 1;
    if (wid == 0) {
      if (it + 1 < KIT) {               // stage next slab into the other buffer
        tdm_load_tile(Abase + (it + 1) * BK, cur ? ldsA0 : ldsA1, BK, BM, Dsz);
        tdm_load_tile(Bbase + (it + 1) * BK, cur ? ldsB0 : ldsB1, BK, BN, Dsz);
        __builtin_amdgcn_s_wait_tensorcnt(2);   // current slab complete
      } else {
        __builtin_amdgcn_s_wait_tensorcnt(0);
      }
    }
    __syncthreads();                    // cur slab visible to all waves

#pragma unroll
    for (int kh = 0; kh < 2; ++kh) {    // two 32-deep K halves
      v16bf af[4], bfr[2];
#pragma unroll
      for (int mi = 0; mi < 4; ++mi)
        af[mi] = load_frag(&As[cur][waveM * 64 + mi * 16 + l][kh * 32], lh);
#pragma unroll
      for (int ni = 0; ni < 2; ++ni)
        bfr[ni] = load_frag(&Bs[cur][waveN * 32 + ni * 16 + l][kh * 32], lh);
#pragma unroll
      for (int mi = 0; mi < 4; ++mi)
#pragma unroll
        for (int ni = 0; ni < 2; ++ni)
          acc[mi][ni] = wmma_bf16(af[mi], bfr[ni], acc[mi][ni]);
    }
    __syncthreads();                    // done reading cur before it is re-staged
  }

  // epilogue: C element r <-> M = r + 8*lh, N = lane&15
#pragma unroll
  for (int mi = 0; mi < 4; ++mi) {
#pragma unroll
    for (int ni = 0; ni < 2; ++ni) {
      int n = nbase + waveN * 32 + ni * 16 + l;
      float bias = biasv[n];
      if constexpr (MODE == 0) {
        int g = n >> 10, n2 = n & 1023;
        int hh = n2 >> 6, dd = n2 & 63;
#pragma unroll
        for (int r = 0; r < 8; ++r) {
          int m = mbase + waveM * 64 + mi * 16 + r + lh * 8;
          int bb = m >> 6, tt = m & 63;
          uint16_t hv = (uint16_t)f2bf(acc[mi][ni][r] + bias);
          size_t head = ((size_t)bb * Hn + hh) * (size_t)(Tsz * DK);
          if (g == 0)      qws [head + (size_t)tt * DK + dd] = hv;
          else if (g == 1) kws [head + (size_t)tt * DK + dd] = hv;
          else             vtws[head + (size_t)dd * Tsz + tt] = hv;
        }
      } else {
#pragma unroll
        for (int r = 0; r < 8; ++r) {
          int m = mbase + waveM * 64 + mi * 16 + r + lh * 8;
          outp[(size_t)m * Dsz + n] = acc[mi][ni][r] + bias;
        }
      }
    }
  }
}

// ---------------- fused attention core: one block per (b,h) ----------------
__global__ __launch_bounds__(128) void k_attn(
    const uint16_t* __restrict__ qws, const uint16_t* __restrict__ kws,
    const uint16_t* __restrict__ vtws, const float* __restrict__ btab,
    const int* __restrict__ mask, uint16_t* __restrict__ ctx) {
  __shared__ __align__(16) uint16_t Ks [64][LSTRIDE];
  __shared__ __align__(16) uint16_t Vts[64][LSTRIDE];
  __shared__ __align__(16) uint16_t Ps [64][LSTRIDE];

  const int tid = threadIdx.x;
  const int wid = tid >> 5, lane = tid & 31;
  const int l = lane & 15, lh = lane >> 4;
  const int bh = blockIdx.x;
  const int b = bh >> 4, h = bh & 15;

  if (wid == 0) {                       // TDM-stage K and V^T tiles (8KB each)
    tdm_load_tile(kws  + (size_t)bh * (Tsz * DK), (uint32_t)(uintptr_t)&Ks[0][0],  DK, Tsz, DK);
    tdm_load_tile(vtws + (size_t)bh * (Tsz * DK), (uint32_t)(uintptr_t)&Vts[0][0], DK, Tsz, DK);
    __builtin_amdgcn_s_wait_tensorcnt(0);
  }
  __syncthreads();

  const int qbase = wid * 16;
  v8f zero = {0.f, 0.f, 0.f, 0.f, 0.f, 0.f, 0.f, 0.f};
  v8f s[4] = {zero, zero, zero, zero};

  // S = Q K^T : A rows = q (this wave's 16), B lane = key index, elements = feature
  const uint16_t* qrow = qws + (size_t)bh * (Tsz * DK) + (size_t)(qbase + l) * DK;
#pragma unroll
  for (int kk = 0; kk < DK; kk += 32) {
    v16bf aq = load_frag(qrow + kk, lh);
#pragma unroll
    for (int t = 0; t < 4; ++t) {
      v16bf bk = load_frag(&Ks[t * 16 + l][kk], lh);
      s[t] = wmma_bf16(aq, bk, s[t]);
    }
  }

  int mk[4];
#pragma unroll
  for (int t = 0; t < 4; ++t) mk[t] = mask[b * Tsz + t * 16 + l];

  float sv[8][4];
#pragma unroll
  for (int r = 0; r < 8; ++r) {
    int q = qbase + r + lh * 8;
    const float* brow = btab + ((size_t)h * Tsz + q) * Tsz;
#pragma unroll
    for (int t = 0; t < 4; ++t) {
      float v = s[t][r] * 0.125f + brow[t * 16 + l];     // 1/sqrt(64)
      if (mk[t] == 0) v = -10000.0f;
      v = fminf(fmaxf(v, -50.0f), 50.0f);
      sv[r][t] = v;
    }
  }

  // softmax: each row lives on the 16 lanes sharing lh (width-16 shuffles)
#pragma unroll
  for (int r = 0; r < 8; ++r) {
    float m = fmaxf(fmaxf(sv[r][0], sv[r][1]), fmaxf(sv[r][2], sv[r][3]));
    m = fmaxf(m, __shfl_xor(m, 1, 16));
    m = fmaxf(m, __shfl_xor(m, 2, 16));
    m = fmaxf(m, __shfl_xor(m, 4, 16));
    m = fmaxf(m, __shfl_xor(m, 8, 16));
    float sum = 0.f;
#pragma unroll
    for (int t = 0; t < 4; ++t) { float e = __expf(sv[r][t] - m); sv[r][t] = e; sum += e; }
    sum += __shfl_xor(sum, 1, 16);
    sum += __shfl_xor(sum, 2, 16);
    sum += __shfl_xor(sum, 4, 16);
    sum += __shfl_xor(sum, 8, 16);
    float inv = 1.0f / sum;
    int q = qbase + r + lh * 8;
#pragma unroll
    for (int t = 0; t < 4; ++t)
      Ps[q][t * 16 + l] = (uint16_t)f2bf(sv[r][t] * inv);
  }
  __syncthreads();

  // ctx = P V : A rows = q, B lane = feature d (from V^T), elements = key index
  v8f c[4] = {zero, zero, zero, zero};
#pragma unroll
  for (int kk = 0; kk < Tsz; kk += 32) {
    v16bf ap = load_frag(&Ps[qbase + l][kk], lh);
#pragma unroll
    for (int t = 0; t < 4; ++t) {
      v16bf bv = load_frag(&Vts[t * 16 + l][kk], lh);
      c[t] = wmma_bf16(ap, bv, c[t]);
    }
  }

  uint16_t* crow = ctx + (size_t)b * Tsz * Dsz + h * DK;   // (b,t,h*64+d)
#pragma unroll
  for (int r = 0; r < 8; ++r) {
    int q = qbase + r + lh * 8;
#pragma unroll
    for (int t = 0; t < 4; ++t)
      crow[(size_t)q * Dsz + t * 16 + l] = (uint16_t)f2bf(c[t][r]);
  }
}

} // anonymous namespace

extern "C" void kernel_launch(void* const* d_in, const int* in_sizes, int n_in,
                              void* d_out, int out_size, void* d_ws, size_t ws_size,
                              hipStream_t stream) {
  const float* x   = (const float*)d_in[0];
  const int*   msk = (const int*)d_in[1];
  const float* Wq  = (const float*)d_in[2];
  const float* Wk  = (const float*)d_in[3];
  const float* Wv  = (const float*)d_in[4];
  const float* Wo  = (const float*)d_in[5];
  const float* bq  = (const float*)d_in[6];
  const float* bk  = (const float*)d_in[7];
  const float* bv  = (const float*)d_in[8];
  const float* bo  = (const float*)d_in[9];
  const float* rel = (const float*)d_in[10];
  float* out = (float*)d_out;

  char* p = (char*)d_ws;
  auto take = [&](size_t bytes) { char* r = p; p += (bytes + 255) & ~(size_t)255; return r; };
  const size_t qkvElems = (size_t)Bsz * Hn * Tsz * DK;          // 33.5M
  uint16_t* qws   = (uint16_t*)take(qkvElems * 2);
  uint16_t* kws   = (uint16_t*)take(qkvElems * 2);
  uint16_t* vtws  = (uint16_t*)take(qkvElems * 2);
  uint16_t* ctxws = (uint16_t*)take((size_t)Mtot * Dsz * 2);
  uint16_t* xbf   = (uint16_t*)take((size_t)Mtot * Dsz * 2);
  uint16_t* wqkvT = (uint16_t*)take((size_t)Nqkv * Dsz * 2);
  uint16_t* woT   = (uint16_t*)take((size_t)Dsz * Dsz * 2);
  float*    bqkv  = (float*)take((size_t)Nqkv * 4);
  float*    btab  = (float*)take((size_t)Hn * Tsz * Tsz * 4);

  k_prep_x   <<<(Mtot * Dsz + 255) / 256, 256, 0, stream>>>(x, xbf);
  k_prep_wqkv<<<(Nqkv * Dsz + 255) / 256, 256, 0, stream>>>(Wq, Wk, Wv, wqkvT);
  k_prep_wo  <<<(Dsz * Dsz + 255) / 256, 256, 0, stream>>>(Wo, woT);
  k_prep_bias<<<(Nqkv + 255) / 256, 256, 0, stream>>>(bq, bk, bv, bqkv);
  k_bias_tab <<<(Hn * Tsz * Tsz + 255) / 256, 256, 0, stream>>>(rel, btab);

  k_gemm<0><<<dim3(Nqkv / BN, Mtot / BM), 256, 0, stream>>>(xbf, wqkvT, bqkv,
                                                            qws, kws, vtws, nullptr);
  k_attn<<<Bsz * Hn, 128, 0, stream>>>(qws, kws, vtws, btab, msk, ctxws);
  k_gemm<1><<<dim3(Dsz / BN, Mtot / BM), 256, 0, stream>>>(ctxws, woT, bo,
                                                           nullptr, nullptr, nullptr, out);

  (void)in_sizes; (void)n_in; (void)out_size; (void)ws_size;
}
